// MotorHead_42348377539078
// MI455X (gfx1250) — matrix operand order repaired and checked by reading, not compiled, not run on hardware
//
#include <hip/hip_runtime.h>
#include <hip/hip_bf16.h>

typedef __attribute__((ext_vector_type(16))) __bf16 v16bf;
typedef __attribute__((ext_vector_type(8)))  __bf16 v8bf;
typedef __attribute__((ext_vector_type(8)))  float  v8f;

#define BATCH 4
#define L_SEQ 2048
#define NEMBD 768
#define NHID  32
#define NBINS 8
#define NV    64

union ABOp { v16bf v16; v8bf v8[2]; };

// ---------------------------------------------------------------------------
// Zero the scalar output (harness poisons d_out before timing).
// ---------------------------------------------------------------------------
__global__ void mh_zero_out(float* out) {
  if (threadIdx.x == 0) out[0] = 0.0f;
}

// ---------------------------------------------------------------------------
// Phase 1a: next-occurrence table.
// tok[(b*NV+v)*L + i] = min k >= i with targets[b,k] == task_tokens[v], else L.
// Replaces the reference's O(B*L*L) segment_min with an O(B*V*L) backward scan.
// ---------------------------------------------------------------------------
__global__ void mh_build_tok(const int* __restrict__ targets,
                             const int* __restrict__ task_tokens,
                             int* __restrict__ tokws) {
  int t = blockIdx.x * blockDim.x + threadIdx.x;
  if (t >= BATCH * NV) return;
  int b = t / NV, v = t - (t / NV) * NV;
  int tt = task_tokens[v];
  const int* tb = targets + (size_t)b * L_SEQ;
  int* row = tokws + (size_t)t * L_SEQ;
  int nxt = L_SEQ;  // sentinel: no event
  for (int k = L_SEQ - 1; k >= 0; --k) {
    if (tb[k] == tt) nxt = k;
    row[k] = nxt;
  }
}

// ---------------------------------------------------------------------------
// Phase 1b: pre-swizzle W1 (768x256) and W2 (32x64) into bf16 WMMA B-operand
// order.  Per (kchunk c, ntile t): 32 lanes x 16 bf16 contiguous, where
// lane = n_local + 16*(K>>4), element e = K & 15  (contiguous k order, same as the A loads).
// Hot loop then reads each B operand as 32 contiguous bytes per lane.
// ---------------------------------------------------------------------------
__global__ void mh_swizzle_w(const float* __restrict__ W1,
                             const float* __restrict__ W2,
                             __bf16* __restrict__ W1sw,
                             __bf16* __restrict__ W2sw) {
  int idx = blockIdx.x * blockDim.x + threadIdx.x;
  if (idx < NEMBD * 256) {
    int k = idx >> 8, n = idx & 255;
    int c = k >> 5, kk = k & 31;
    int t = n >> 4, nl = n & 15;
    int kh = kk >> 4, e = kk & 15;
    W1sw[((size_t)(c * 16 + t) * 32 + nl + 16 * kh) * 16 + e] = (__bf16)W1[idx];
  }
  if (idx < NHID * NV) {
    int k = idx >> 6, n = idx & 63;
    int t = n >> 4, nl = n & 15;
    int kh = k >> 4, e = k & 15;
    W2sw[((size_t)t * 32 + nl + 16 * kh) * 16 + e] = (__bf16)W2[idx];
  }
}

// ---------------------------------------------------------------------------
// Phase 2: fused  h@W1 -> hh@W2 -> survival loss  per 16-token tile.
// Grid: BATCH * (L/16) = 512 blocks, 256 threads (8 waves).
//   A operands: row-major bf16 in LDS; two ds_load_b128 per operand
//   (row-major B128 loads give contiguous k = 16*kh + s, same order as B).
//   B operands: two coalesced global_load_b128 from pre-swizzled W1sw/W2sw.
//   Wave w's GEMM1 output (hh cols 32w..32w+31) is exactly its GEMM2 A tile
//   (bin n = w): stored wave-private in LDS, no inter-wave barrier needed.
// ---------------------------------------------------------------------------
__global__ __launch_bounds__(256) void mh_fused_loss(
    const float* __restrict__ h,     const float* __restrict__ age,
    const float* __restrict__ tage,  const __bf16* __restrict__ W1sw,
    const __bf16* __restrict__ W2sw, const float* __restrict__ tbins,
    const int*  __restrict__ tokws,  float* __restrict__ out) {

  __shared__ __align__(32) __bf16 Ash[16 * NEMBD];   // 24 KB row-major bf16
  __shared__ __align__(32) __bf16 HHw[8][16 * 32];   //  8 KB, per-wave hh tile
  __shared__ float  TTEsh[16 * NV];                  //  4 KB
  __shared__ unsigned char NOEVsh[16 * NV];
  __shared__ float  LASTCsh[16];
  __shared__ float  REDsh[256];

  const int tid  = threadIdx.x;
  const int lane = tid & 31;
  const int wave = tid >> 5;
  const int b    = blockIdx.x >> 7;            // 128 tiles per batch row
  const int i0   = (blockIdx.x & 127) << 4;

  // Stage h tile (coalesced fp32 load, bf16 row-major store to LDS)
  for (int idx = tid; idx < 16 * NEMBD; idx += 256) {
    int m = idx / NEMBD, k = idx - m * NEMBD;
    Ash[idx] = (__bf16)h[((size_t)(b * L_SEQ + i0 + m)) * NEMBD + k];
  }

  // Precompute tte / no_event / censor for this tile
  const float lastTA = tage[b * L_SEQ + L_SEQ - 1];
  for (int idx = tid; idx < 16 * NV; idx += 256) {
    int m = idx >> 6, v = idx & 63;
    int tk = tokws[((size_t)(b * NV + v)) * L_SEQ + i0 + m];
    NOEVsh[idx] = (tk >= L_SEQ) ? 1 : 0;
    tk = tk < (L_SEQ - 1) ? tk : (L_SEQ - 1);
    TTEsh[idx] = tage[b * L_SEQ + tk] - age[b * L_SEQ + i0 + m];
  }
  if (tid < 16) LASTCsh[tid] = lastTA - age[b * L_SEQ + i0 + tid];
  __syncthreads();

  const int mrow = lane & 15;   // M row (A) / N col (B,C) per Sec 7.12.2
  const int kh   = lane >> 4;   // K-half selector

  // ---------------- GEMM1: [16x768] x [768x256] ----------------
  v8f c0 = {}, c1 = {};
  {
    const __bf16* arow = &Ash[mrow * NEMBD + kh * 16];
    const __bf16* bptr = W1sw + ((size_t)(2 * wave) * 32 + lane) * 16;
    for (int c = 0; c < 24; ++c) {
      ABOp a, b0, b1;
      a.v8[0] = *(const v8bf*)(arow + c * 32);
      a.v8[1] = *(const v8bf*)(arow + c * 32 + 8);
      b0.v16 = *(const v16bf*)(bptr + (size_t)c * 16 * 512);
      b1.v16 = *(const v16bf*)(bptr + (size_t)c * 16 * 512 + 512);
      c0 = __builtin_amdgcn_wmma_f32_16x16x32_bf16(false, a.v16, false, b0.v16,
                                                   (short)0, c0, false, false);
      c1 = __builtin_amdgcn_wmma_f32_16x16x32_bf16(false, a.v16, false, b1.v16,
                                                   (short)0, c1, false, false);
    }
  }

  // Park hh tile (wave-private) as row-major bf16: rows m, 32 local cols.
  {
    __bf16* hw = HHw[wave];
#pragma unroll
    for (int r = 0; r < 8; ++r) {
      int m = r + 8 * kh;                 // C layout: VGPR r, lane>>4
      hw[m * 32 + (lane & 15)]      = (__bf16)c0[r];
      hw[m * 32 + 16 + (lane & 15)] = (__bf16)c1[r];
    }
  }

  // ---------------- GEMM2 + loss: bin n = wave ----------------
  const float t0   = tbins[wave], t1 = tbins[wave + 1];
  const float binw = t1 - t0;
  float acc = 0.0f;

  ABOp a2;  // same A for all 4 v-tiles (hh rows x K=32)
  a2.v8[0] = *(const v8bf*)&HHw[wave][mrow * 32 + kh * 16];
  a2.v8[1] = *(const v8bf*)&HHw[wave][mrow * 32 + kh * 16 + 8];

  for (int vt = 0; vt < 4; ++vt) {
    ABOp bb;
    bb.v16 = *(const v16bf*)(W2sw + ((size_t)vt * 32 + lane) * 16);
    v8f c = {};
    c = __builtin_amdgcn_wmma_f32_16x16x32_bf16(false, a2.v16, false, bb.v16,
                                                (short)0, c, false, false);
#pragma unroll
    for (int r = 0; r < 8; ++r) {
      int m = r + 8 * kh;
      int v = vt * 16 + (lane & 15);
      float ll  = c[r];
      float tte = TTEsh[m * NV + v];
      bool  occ = (tte > t0) && (tte <= t1) && (NOEVsh[m * NV + v] == 0);
      float cen = fminf(fmaxf(LASTCsh[m], 0.0f), binw);
      float s   = occ ? tte : cen;
      // loss = mean( exp(ll)*s - ll*occ )
      acc += __expf(ll) * s - (occ ? ll : 0.0f);
    }
  }

  REDsh[tid] = acc;
  __syncthreads();
  for (int off = 128; off > 0; off >>= 1) {
    if (tid < off) REDsh[tid] += REDsh[tid + off];
    __syncthreads();
  }
  if (tid == 0)
    atomicAdd(out, REDsh[0] *
                       (1.0f / (float)((size_t)BATCH * L_SEQ * NBINS * NV)));
}

// ---------------------------------------------------------------------------
extern "C" void kernel_launch(void* const* d_in, const int* in_sizes, int n_in,
                              void* d_out, int out_size, void* d_ws,
                              size_t ws_size, hipStream_t stream) {
  const float* h           = (const float*)d_in[0];
  const float* age         = (const float*)d_in[1];
  const float* tage        = (const float*)d_in[2];
  const int*   targets     = (const int*)d_in[3];
  const float* W1          = (const float*)d_in[4];
  const float* W2          = (const float*)d_in[5];
  const int*   task_tokens = (const int*)d_in[6];
  const float* tbins       = (const float*)d_in[7];
  float* out = (float*)d_out;

  // Workspace layout: tok table (2 MB) | W1sw bf16 (384 KB) | W2sw bf16 (4 KB)
  int*    tokws = (int*)d_ws;
  __bf16* W1sw  = (__bf16*)((char*)d_ws + (size_t)BATCH * NV * L_SEQ * 4);
  __bf16* W2sw  = W1sw + (size_t)NEMBD * 256;

  mh_zero_out<<<1, 32, 0, stream>>>(out);
  mh_build_tok<<<1, 256, 0, stream>>>(targets, task_tokens, tokws);
  mh_swizzle_w<<<(NEMBD * 256 + 255) / 256, 256, 0, stream>>>(W1, W2, W1sw,
                                                              W2sw);
  mh_fused_loss<<<BATCH * (L_SEQ / 16), 256, 0, stream>>>(
      h, age, tage, W1sw, W2sw, tbins, tokws, out);
}
